// variance_network_20830591386409
// MI455X (gfx1250) — compile-verified, matrix operand order checked by reading, not run on hardware
//
#include <hip/hip_runtime.h>
#include <stdint.h>

// ---------------------------------------------------------------------------
// Types for CDNA5 WMMA (wave32): v_wmma_f32_16x16x32_bf16
// ---------------------------------------------------------------------------
typedef __bf16  v16bf __attribute__((ext_vector_type(16)));
typedef float   v8f   __attribute__((ext_vector_type(8)));

union FragBF {
  v16bf v;
  unsigned short h[16];
  uint4 q[2];
};

static __device__ __forceinline__ unsigned short f2bf(float f) {
  union { float f; unsigned u; } x; x.f = f;
  unsigned u = x.u;
  return (unsigned short)((u + 0x7FFFu + ((u >> 16) & 1u)) >> 16);  // RNE
}
static __device__ __forceinline__ float bf2f(unsigned short h) {
  union { unsigned u; float f; } x; x.u = ((unsigned)h) << 16;
  return x.f;
}

// A-fragment (16x32 bf16) lane/element mapping per ISA 7.12.2:
//   lane L<16 : row M=L,    elems e=0..7 -> K=e,   e=8..15 -> K=16+(e-8)
//   lane L>=16: row M=L-16, elems e=0..7 -> K=8+e, e=8..15 -> K=24+(e-8)
// For element (r, k): lane = r + 16*((k>>3)&1); e = (k&7) + ((k>>4)<<3)

#define FEAT  512
#define NPTS  2048
#define NBAT  16
#define MROWS (NBAT * NPTS)   // 32768

// ---------------------------------------------------------------------------
// Pack weights [rows x 512] f32 -> B-fragment-swizzled bf16.
// ---------------------------------------------------------------------------
__global__ __launch_bounds__(256)
void pack_w_kernel(const float* __restrict__ W, int rows,
                   unsigned short* __restrict__ dst)
{
  int t    = blockIdx.x * 256 + threadIdx.x;
  int lane = t & 31;
  int nt   = (t >> 5) & 31;
  int kt   = t >> 10;
  int half = lane >> 4;
  int n    = nt * 16 + (lane & 15);

  unsigned short v[16];
#pragma unroll
  for (int e = 0; e < 16; ++e) {
    int k = kt * 32 + (e & 7) + half * 8 + ((e >> 3) << 4);
    v[e] = (k < rows) ? f2bf(W[(size_t)k * 512 + n]) : (unsigned short)0;
  }
  uint4* o = (uint4*)(dst + (size_t)t * 16);
  o[0] = ((const uint4*)v)[0];
  o[1] = ((const uint4*)v)[1];
}

// ---------------------------------------------------------------------------
// kNN + local covariance -> extra A-fragment tile (bf16, 32 K cols, 12 valid)
// ---------------------------------------------------------------------------
__global__ __launch_bounds__(256)
void knn_extra_kernel(const float* __restrict__ pc,
                      unsigned short* __restrict__ extra)
{
  __shared__ float spx[NPTS], spy[NPTS], spz[NPTS];
  const int b = blockIdx.x;
  const float* p = pc + (size_t)b * NPTS * 3;

  for (int i = threadIdx.x; i < NPTS; i += 256) {
    spx[i] = p[i * 3 + 0];
    spy[i] = p[i * 3 + 1];
    spz[i] = p[i * 3 + 2];
  }
  __syncthreads();

  for (int n = threadIdx.x; n < NPTS; n += 256) {
    float x = spx[n], y = spy[n], z = spz[n];
    float best = -3.4e38f;
    int   bi   = n;
    for (int m = 0; m < NPTS; ++m) {
      float dx = x - spx[m], dy = y - spy[m], dz = z - spz[m];
      float nd = -(dx * dx + dy * dy + dz * dz);
      bool better = (m != n) && (nd > best);
      best = better ? nd : best;
      bi   = better ? m  : bi;
    }
    float qx = spx[bi], qy = spy[bi], qz = spz[bi];
    float feat[12] = { x, y, z,
                       x * qx, x * qy, x * qz,
                       y * qx, y * qy, y * qz,
                       z * qx, z * qy, z * qz };
    int mg = b * NPTS + n;
    int mt = mg >> 4, r = mg & 15;
#pragma unroll
    for (int k = 0; k < 32; ++k) {
      int lane = r + (((k >> 3) & 1) << 4);
      int e    = (k & 7) + ((k >> 4) << 3);
      extra[(size_t)mt * 512 + lane * 16 + e] =
          (k < 12) ? f2bf(feat[k]) : (unsigned short)0;
    }
  }
}

// ---------------------------------------------------------------------------
// Running scalar output -> extra A-fragment tile (K col 0 valid, rest zero)
// ---------------------------------------------------------------------------
__global__ __launch_bounds__(256)
void make_extra_kernel(const float* __restrict__ s,
                       unsigned short* __restrict__ extra)
{
  int t    = blockIdx.x * 256 + threadIdx.x;   // 65536 threads: (mt, lane)
  int lane = t & 31;
  int mt   = t >> 5;
  unsigned short v[16] = {0};
  if (lane < 16) v[0] = f2bf(s[mt * 16 + lane]);   // lane<16: e=0 <-> K=0
  uint4* o = (uint4*)(extra + (size_t)t * 16);
  o[0] = ((const uint4*)v)[0];
  o[1] = ((const uint4*)v)[1];
}

// ---------------------------------------------------------------------------
// Fused residual MLP with software-pipelined WMMA:
//   out[m] = (prev?prev[m]:0) + W2 . relu(h + h@W1 + b1) + b2
//   h = relu([latent(512) | extra(32)] @ W0 + b0)
// 64 rows/block, 8 waves; wave owns 16-row x 256-col strip (16 C fragments).
// B fragments flow through a 4-deep register ring; A double-buffered.
// ---------------------------------------------------------------------------
__global__ __launch_bounds__(256)
void mlp_wmma_kernel(const float* __restrict__ latent,
                     const unsigned short* __restrict__ extra,
                     const unsigned short* __restrict__ w0f,
                     const float* __restrict__ b0,
                     const unsigned short* __restrict__ w1f,
                     const float* __restrict__ b1,
                     const float* __restrict__ w2,
                     const float* __restrict__ b2,
                     const float* __restrict__ prev,
                     float* __restrict__ out)
{
  __shared__ unsigned short lds_h[64 * 528];   // 64 rows x 512 (+16 pad) bf16
  __shared__ float lds_p[256];

  const int t       = threadIdx.x;
  const int lane    = t & 31;
  const int wv      = t >> 5;
  const int strip   = wv & 3;      // 4 row-strips of 16
  const int colHalf = wv >> 2;     // 2 column halves of 256
  const int half    = lane >> 4;
  const int ln      = lane & 15;
  const int m0      = blockIdx.x * 64;
  const int rowA    = m0 + strip * 16 + ln;       // global A row this lane
  const int rowL    = strip * 16 + ln;            // local A row (LDS)
  const int rbase   = strip * 16 + half * 8;      // C-layout local row base

  v8f acc[16];
#pragma unroll
  for (int i = 0; i < 16; ++i)
#pragma unroll
    for (int j = 0; j < 8; ++j) acc[i][j] = 0.0f;

  FragBF a0, a1, bb0, bb1, bb2, bb3;

  // --- fragment loaders ---
  auto loadB = [&](FragBF& f, const unsigned short* wf, int frag) {
    const unsigned short* bp = wf + (size_t)frag * 512 + (size_t)lane * 16;
    f.q[0] = *(const uint4*)(bp);
    f.q[1] = *(const uint4*)(bp + 8);
  };
  auto loadA0 = [&](FragBF& f, int kt) {        // layer-0 A: latent f32 / extra
    if (kt < 16) {
      const float* src = latent + (size_t)rowA * 512 + kt * 32 + half * 8;
      float4 f0 = *(const float4*)(src);
      float4 f1 = *(const float4*)(src + 4);
      float4 f2 = *(const float4*)(src + 16);
      float4 f3 = *(const float4*)(src + 20);
      float tf[16] = { f0.x, f0.y, f0.z, f0.w, f1.x, f1.y, f1.z, f1.w,
                       f2.x, f2.y, f2.z, f2.w, f3.x, f3.y, f3.z, f3.w };
#pragma unroll
      for (int e = 0; e < 16; ++e) f.h[e] = f2bf(tf[e]);
    } else {
      const unsigned short* eb =
          extra + ((size_t)(blockIdx.x * 4 + strip) * 512 + lane * 16);
      f.q[0] = *(const uint4*)(eb);
      f.q[1] = *(const uint4*)(eb + 8);
    }
  };
  auto loadA1 = [&](FragBF& f, int kt) {        // layer-1 A: h0 bf16 from LDS
    const unsigned short* ab = lds_h + rowL * 528 + kt * 32 + half * 8;
    f.q[0] = *(const uint4*)(ab);
    f.q[1] = *(const uint4*)(ab + 16);
  };

#define BBSLOT(i) ((i) == 0 ? bb0 : ((i) == 1 ? bb1 : ((i) == 2 ? bb2 : bb3)))

  // 16 WMMAs for one k-tile: consume ring slot nt&3, refill it for step +4.
  // Ring over-reads <=4 fragments past the weight buffer (workspace-backed).
  auto body0 = [&](int kt, FragBF& aUse, FragBF& aPre) {
    if (kt + 1 < 17) loadA0(aPre, kt + 1);
    __builtin_prefetch(w0f + ((size_t)(kt + 2) * 32 + colHalf * 16) * 512 +
                           (size_t)lane * 16, 0, 0);
#pragma unroll
    for (int nt = 0; nt < 16; ++nt) {
      acc[nt] = __builtin_amdgcn_wmma_f32_16x16x32_bf16(
          false, aUse.v, false, BBSLOT(nt & 3).v, (short)0, acc[nt],
          false, false);
      int pkt = kt + ((nt + 4) >> 4);
      int pnt = (nt + 4) & 15;
      loadB(BBSLOT(nt & 3), w0f, pkt * 32 + colHalf * 16 + pnt);
    }
  };
  auto body1 = [&](int kt, FragBF& aUse, FragBF& aPre) {
    if (kt + 1 < 16) loadA1(aPre, kt + 1);
    __builtin_prefetch(w1f + ((size_t)(kt + 2) * 32 + colHalf * 16) * 512 +
                           (size_t)lane * 16, 0, 0);
#pragma unroll
    for (int nt = 0; nt < 16; ++nt) {
      acc[nt] = __builtin_amdgcn_wmma_f32_16x16x32_bf16(
          false, aUse.v, false, BBSLOT(nt & 3).v, (short)0, acc[nt],
          false, false);
      int pkt = kt + ((nt + 4) >> 4);
      int pnt = (nt + 4) & 15;
      loadB(BBSLOT(nt & 3), w1f, pkt * 32 + colHalf * 16 + pnt);
    }
  };

  // ---------------- Layer 0: [latent | extra] @ W0, KT = 17 ----------------
  loadA0(a0, 0);
  loadB(bb0, w0f, colHalf * 16 + 0);
  loadB(bb1, w0f, colHalf * 16 + 1);
  loadB(bb2, w0f, colHalf * 16 + 2);
  loadB(bb3, w0f, colHalf * 16 + 3);
#pragma unroll 1
  for (int kt = 0; kt < 16; kt += 2) {
    body0(kt,     a0, a1);
    body0(kt + 1, a1, a0);
  }
  body0(16, a0, a1);   // extra k-tile (A prefetched into a0 during kt=15)

  // bias + ReLU -> h0 into LDS (bf16)
#pragma unroll
  for (int nt = 0; nt < 16; ++nt) {
    int n = colHalf * 256 + nt * 16 + ln;
    float bias = b0[n];
#pragma unroll
    for (int g = 0; g < 8; ++g) {
      float v = acc[nt][g] + bias;
      v = v > 0.0f ? v : 0.0f;
      lds_h[(rbase + g) * 528 + n] = f2bf(v);
    }
  }
  __syncthreads();

  // ---------------- Layer 1 (residual): h1 = relu(h0 + h0@W1 + b1) ---------
#pragma unroll
  for (int i = 0; i < 16; ++i)
#pragma unroll
    for (int j = 0; j < 8; ++j) acc[i][j] = 0.0f;

  loadA1(a0, 0);
  loadB(bb0, w1f, colHalf * 16 + 0);
  loadB(bb1, w1f, colHalf * 16 + 1);
  loadB(bb2, w1f, colHalf * 16 + 2);
  loadB(bb3, w1f, colHalf * 16 + 3);
#pragma unroll 1
  for (int kt = 0; kt < 16; kt += 2) {
    body1(kt,     a0, a1);
    body1(kt + 1, a1, a0);
  }

  // residual + bias + ReLU (read h0 before anyone overwrites LDS)
#pragma unroll
  for (int nt = 0; nt < 16; ++nt) {
    int n = colHalf * 256 + nt * 16 + ln;
    float bias = b1[n];
#pragma unroll
    for (int g = 0; g < 8; ++g) {
      float h0v = bf2f(lds_h[(rbase + g) * 528 + n]);
      float v = h0v + acc[nt][g] + bias;
      acc[nt][g] = v > 0.0f ? v : 0.0f;
    }
  }
  __syncthreads();
#pragma unroll
  for (int nt = 0; nt < 16; ++nt) {
    int n = colHalf * 256 + nt * 16 + ln;
#pragma unroll
    for (int g = 0; g < 8; ++g)
      lds_h[(rbase + g) * 528 + n] = f2bf(acc[nt][g]);
  }
  __syncthreads();

  // ---------------- Layer 2: 512 -> 1 GEMV from LDS ------------------------
  {
    int row  = t & 63;
    int part = t >> 6;                 // 4-way K split
    const unsigned short* hr = lds_h + row * 528 + part * 128;
    const float* wp = w2 + part * 128;
    float s = 0.0f;
#pragma unroll 4
    for (int k = 0; k < 128; ++k) s += bf2f(hr[k]) * wp[k];
    lds_p[row * 4 + part] = s;
  }
  __syncthreads();
  if (t < 64) {
    float s = lds_p[t * 4] + lds_p[t * 4 + 1] + lds_p[t * 4 + 2] +
              lds_p[t * 4 + 3] + b2[0];
    int m = m0 + t;
    out[m] = s + (prev ? prev[m] : 0.0f);
  }
#undef BBSLOT
}

// ---------------------------------------------------------------------------
// Host-side launch
// ---------------------------------------------------------------------------
extern "C" void kernel_launch(void* const* d_in, const int* in_sizes, int n_in,
                              void* d_out, int out_size, void* d_ws, size_t ws_size,
                              hipStream_t stream)
{
  (void)in_sizes; (void)n_in; (void)out_size; (void)ws_size;

  const float* latent = (const float*)d_in[0];
  const float* pc     = (const float*)d_in[1];
  const float* W[3][3];
  const float* Bv[3][3];
  int idx = 2;
  for (int s = 0; s < 3; ++s)
    for (int j = 0; j < 3; ++j) {
      W[s][j]  = (const float*)d_in[idx++];
      Bv[s][j] = (const float*)d_in[idx++];
    }

  char* ws = (char*)d_ws;
  size_t off = 0;
  unsigned short* extra = (unsigned short*)(ws + off); off += (size_t)2048 * 512 * 2; // 2 MB
  unsigned short* w0f[3];
  for (int i = 0; i < 3; ++i) { w0f[i] = (unsigned short*)(ws + off); off += (size_t)17 * 32 * 512 * 2; }
  unsigned short* w1f[3];
  for (int i = 0; i < 3; ++i) { w1f[i] = (unsigned short*)(ws + off); off += (size_t)16 * 32 * 512 * 2; }
  float* tmp1 = (float*)(ws + off); off += (size_t)MROWS * 4;
  float* tmp2 = (float*)(ws + off); off += (size_t)MROWS * 4;

  // Pack weights (tiny; weights stay L2-resident afterwards)
  for (int s = 0; s < 3; ++s) {
    int rows0 = (s == 0) ? 524 : 513;
    pack_w_kernel<<<17 * 4, 256, 0, stream>>>(W[s][0], rows0, w0f[s]);
    pack_w_kernel<<<16 * 4, 256, 0, stream>>>(W[s][1], 512,   w1f[s]);
  }

  // kNN local covariance -> extra tile for MLP 1
  knn_extra_kernel<<<NBAT, 256, 0, stream>>>(pc, extra);

  // MLP 1
  mlp_wmma_kernel<<<MROWS / 64, 256, 0, stream>>>(
      latent, extra, w0f[0], Bv[0][0], w1f[0], Bv[0][1], W[0][2], Bv[0][2],
      nullptr, tmp1);

  // MLP 2 (extra = out1, residual prev = out1)
  make_extra_kernel<<<256, 256, 0, stream>>>(tmp1, extra);
  mlp_wmma_kernel<<<MROWS / 64, 256, 0, stream>>>(
      latent, extra, w0f[1], Bv[1][0], w1f[1], Bv[1][1], W[1][2], Bv[1][2],
      tmp1, tmp2);

  // MLP 3 -> final output
  make_extra_kernel<<<256, 256, 0, stream>>>(tmp2, extra);
  mlp_wmma_kernel<<<MROWS / 64, 256, 0, stream>>>(
      latent, extra, w0f[2], Bv[2][0], w1f[2], Bv[2][1], W[2][2], Bv[2][2],
      tmp2, (float*)d_out);
}
